// Tokenizer_66924180407139
// MI455X (gfx1250) — matrix operand order repaired and checked by reading, not compiled, not run on hardware
//
#include <hip/hip_runtime.h>
#include <stdint.h>

#define D          256     // embed dim
#define NCHUNK     16      // codebook codes per chunk (one 16x16 WMMA tile of N)
#define ROWP       264     // padded LDS row length in ushorts (264*2=528B, conflict-free)
#define MB         128     // rows per block (8 waves * 16 rows)

typedef __attribute__((ext_vector_type(16))) __bf16       v16bf;
typedef __attribute__((ext_vector_type(8)))  float        v8f;
typedef __attribute__((ext_vector_type(8)))  unsigned int v8u;

// f32 -> bf16 (round-to-nearest-even), bit manipulation (no __bf16 arithmetic needed)
__device__ __forceinline__ unsigned f2bf(float x) {
  unsigned u = __float_as_uint(x);
  u += 0x7fffu + ((u >> 16) & 1u);
  return u >> 16;
}
__device__ __forceinline__ float bf2f(unsigned h) {
  return __uint_as_float(h << 16);
}

// ---------------- prep: split codebook into bf16 hi/lo, compute squared norms ---------
__global__ void cb_prep_kernel(const float* __restrict__ cb,
                               unsigned short* __restrict__ cb_hi,
                               unsigned short* __restrict__ cb_lo,
                               float* __restrict__ b2) {
  const int row = blockIdx.x;
  const int col = threadIdx.x;
  const size_t idx = (size_t)row * D + col;
  float x = cb[idx];
  unsigned hi = f2bf(x);
  cb_hi[idx] = (unsigned short)hi;
  cb_lo[idx] = (unsigned short)f2bf(x - bf2f(hi));

  float v = x * x;
  #pragma unroll
  for (int off = 16; off >= 1; off >>= 1) v += __shfl_xor(v, off, 32);
  __shared__ float part[8];
  const int wave = threadIdx.x >> 5;
  if ((threadIdx.x & 31) == 0) part[wave] = v;
  __syncthreads();
  if (threadIdx.x == 0) {
    float s = 0.f;
    #pragma unroll
    for (int i = 0; i < 8; ++i) s += part[i];
    b2[row] = s;
  }
}

// ---------------- async staging of one 16-code hi/lo chunk into LDS -------------------
__device__ __forceinline__ void stage_chunk(const unsigned short* __restrict__ cb_hi,
                                            const unsigned short* __restrict__ cb_lo,
                                            unsigned short* lds_hi,
                                            unsigned short* lds_lo,
                                            int cbase, int tid) {
  // 16 codes * 512B (hi) = 512 x 16B segments; 256 threads -> 2 segs each, for hi and lo
  #pragma unroll
  for (int rr = 0; rr < 2; ++rr) {
    const int r    = tid + rr * 256;
    const int code = r >> 5;       // 32 x 16B segments per code row
    const int off  = r & 31;
    const unsigned short* gh = cb_hi + (size_t)(cbase + code) * D + off * 8;
    const unsigned short* gl = cb_lo + (size_t)(cbase + code) * D + off * 8;
    unsigned lh = (unsigned)(unsigned long long)(lds_hi + code * ROWP + off * 8);
    unsigned ll = (unsigned)(unsigned long long)(lds_lo + code * ROWP + off * 8);
    asm volatile("global_load_async_to_lds_b128 %0, %1, off"
                 :: "v"(lh), "v"((unsigned long long)gh) : "memory");
    asm volatile("global_load_async_to_lds_b128 %0, %1, off"
                 :: "v"(ll), "v"((unsigned long long)gl) : "memory");
  }
}

// ---------------- main: WMMA distance + running argmin --------------------------------
__global__ __launch_bounds__(256, 2)
void tok_argmin_kernel(const float* __restrict__ rep,
                       const unsigned short* __restrict__ cb_hi,
                       const unsigned short* __restrict__ cb_lo,
                       const float* __restrict__ b2,
                       int* __restrict__ out,
                       int K) {
  __shared__ alignas(16) unsigned short smem[2][2][NCHUNK * ROWP];

  const int tid  = threadIdx.x;
  const int wave = tid >> 5;
  const int lane = tid & 31;
  const int half = lane >> 4;       // 0: lanes 0-15, 1: lanes 16-31
  const int l16  = lane & 15;
  const int kbase = half * 8;       // A-fragment K offset per lane half (ISA 16-bit A layout)
  const size_t row = (size_t)blockIdx.x * MB + wave * 16 + l16;

  // ---- load this wave's 16x256 rep tile, split into bf16 hi/lo A fragments (regs) ----
  v8u ah[8], al[8];
  #pragma unroll
  for (int s = 0; s < 8; ++s) {
    const float* p = rep + row * D + s * 32 + kbase;
    float4 q0 = *(const float4*)(p);
    float4 q1 = *(const float4*)(p + 4);
    float4 q2 = *(const float4*)(p + 16);
    float4 q3 = *(const float4*)(p + 20);
    float xs[16] = {q0.x, q0.y, q0.z, q0.w, q1.x, q1.y, q1.z, q1.w,
                    q2.x, q2.y, q2.z, q2.w, q3.x, q3.y, q3.z, q3.w};
    v8u h, l;
    #pragma unroll
    for (int j = 0; j < 8; ++j) {
      unsigned h0 = f2bf(xs[2 * j]);
      unsigned h1 = f2bf(xs[2 * j + 1]);
      unsigned l0 = f2bf(xs[2 * j]     - bf2f(h0));
      unsigned l1 = f2bf(xs[2 * j + 1] - bf2f(h1));
      h[j] = h0 | (h1 << 16);
      l[j] = l0 | (l1 << 16);
    }
    ah[s] = h;
    al[s] = l;
  }

  float best[8];
  int   bidx[8];
  #pragma unroll
  for (int v = 0; v < 8; ++v) { best[v] = __builtin_inff(); bidx[v] = 0; }

  const int nchunks = K / NCHUNK;
  stage_chunk(cb_hi, cb_lo, smem[0][0], smem[0][1], 0, tid);

  for (int c = 0; c < nchunks; ++c) {
    const int buf   = c & 1;
    const int cbase = c * NCHUNK;

    asm volatile("s_wait_asynccnt 0" ::: "memory");   // chunk c resident in LDS
    __syncthreads();                                  // visible to all waves
    if (c + 1 < nchunks)                              // overlap next chunk's DMA
      stage_chunk(cb_hi, cb_lo, smem[buf ^ 1][0], smem[buf ^ 1][1],
                  cbase + NCHUNK, tid);

    const float b2v = b2[cbase + l16];                // in flight during LDS loads
    const unsigned short* lh = smem[buf][0] + l16 * ROWP + half * 8;
    const unsigned short* ll = smem[buf][1] + l16 * ROWP + half * 8;

    // ---- preload ALL B fragments for this chunk: 32 back-to-back ds_load_b128 ----
    v8u bh[8], bl[8];
    #pragma unroll
    for (int s = 0; s < 8; ++s) {
      uint4 h0 = *(const uint4*)(lh + s * 32);
      uint4 h1 = *(const uint4*)(lh + s * 32 + 16);
      uint4 g0 = *(const uint4*)(ll + s * 32);
      uint4 g1 = *(const uint4*)(ll + s * 32 + 16);
      bh[s] = v8u{h0.x, h0.y, h0.z, h0.w, h1.x, h1.y, h1.z, h1.w};
      bl[s] = v8u{g0.x, g0.y, g0.z, g0.w, g1.x, g1.y, g1.z, g1.w};
    }

    // Hard scheduling fence: loads above may NOT be sunk into the WMMA burst below.
    // Forces all 16 B fragments simultaneously live -> distinct VGPRs, single wait.
#if __has_builtin(__builtin_amdgcn_sched_barrier)
    __builtin_amdgcn_sched_barrier(0);
#endif

    // ---- 24 WMMAs as 3 independent accumulation chains (no serial RAW on C) ----
    v8f acc0 = {}, acc1 = {}, acc2 = {};
    #pragma unroll
    for (int s = 0; s < 8; ++s) {
      v16bf A  = __builtin_bit_cast(v16bf, ah[s]);
      v16bf Al = __builtin_bit_cast(v16bf, al[s]);
      v16bf Bh = __builtin_bit_cast(v16bf, bh[s]);
      v16bf Bl = __builtin_bit_cast(v16bf, bl[s]);
      acc0 = __builtin_amdgcn_wmma_f32_16x16x32_bf16(false, A,  false, Bh, (short)0, acc0, false, false);
      acc1 = __builtin_amdgcn_wmma_f32_16x16x32_bf16(false, Al, false, Bh, (short)0, acc1, false, false);
      acc2 = __builtin_amdgcn_wmma_f32_16x16x32_bf16(false, A,  false, Bl, (short)0, acc2, false, false);
    }

    // dist proxy = b2 - 2*ab  (a2 constant per row -> irrelevant for argmin)
    const int code = cbase + l16;   // this lane's column N = l16
    #pragma unroll
    for (int v = 0; v < 8; ++v) {
      float ab = acc0[v] + (acc1[v] + acc2[v]);
      float d  = __builtin_fmaf(-2.0f, ab, b2v);
      if (d < best[v]) { best[v] = d; bidx[v] = code; }
    }
  }

  // ---- reduce argmin across the 16 lanes of each half (candidates over N) ----
  #pragma unroll
  for (int v = 0; v < 8; ++v) {
    float d = best[v];
    int   bi = bidx[v];
    #pragma unroll
    for (int off = 8; off >= 1; off >>= 1) {
      float od = __shfl_xor(d, off, 32);
      int   oi = __shfl_xor(bi, off, 32);
      if (od < d || (od == d && oi < bi)) { d = od; bi = oi; }
    }
    bidx[v] = bi;
  }
  // C layout: slot v of lanes 0-15 -> row v; of lanes 16-31 -> row v+8
  if (l16 == 0) {
    const size_t orow = (size_t)blockIdx.x * MB + wave * 16 + half * 8;
    #pragma unroll
    for (int v = 0; v < 8; ++v) out[orow + v] = bidx[v];
  }
}

extern "C" void kernel_launch(void* const* d_in, const int* in_sizes, int n_in,
                              void* d_out, int out_size, void* d_ws, size_t ws_size,
                              hipStream_t stream) {
  const float* rep = (const float*)d_in[0];
  const float* cb  = (const float*)d_in[1];
  const int M = in_sizes[0] / D;   // 65536
  const int K = in_sizes[1] / D;   // 8192

  unsigned short* cb_hi = (unsigned short*)d_ws;
  unsigned short* cb_lo = cb_hi + (size_t)K * D;
  float*          b2    = (float*)(cb_lo + (size_t)K * D);
  int*            out   = (int*)d_out;

  cb_prep_kernel<<<K, D, 0, stream>>>(cb, cb_hi, cb_lo, b2);
  tok_argmin_kernel<<<M / MB, 256, 0, stream>>>(rep, cb_hi, cb_lo, b2, out, K);
}
